// AttentionHead_68298569941291
// MI455X (gfx1250) — matrix operand order repaired and checked by reading, not compiled, not run on hardware
//
#include <hip/hip_runtime.h>
#include <hip/hip_bf16.h>

// Attention: q=x@Wq^T+bq, k=..., v=...; softmax(q k^T / sqrt(E)) @ v
// B=4, S=2048, E=1024. All matmuls via v_wmma_f32_16x16x32_bf16, wave32.
// GEMM: 128x128 block tile, BK=64 stages, double-buffered LDS, branch-free
// inner loop (peeled epilogue), gfx1250 async global->LDS staging (ASYNCcnt).

#define BATCH 4
#define SEQ   2048
#define EMB   1024
#define MTOT  (BATCH * SEQ)

#define BM   128
#define BNT  128
#define BK   64
#define LDSP 72   // padded LDS row stride in bf16 elements (144B, 16B aligned)

typedef __attribute__((ext_vector_type(16))) __bf16 v16bf;
typedef __attribute__((ext_vector_type(4)))  __bf16 bf16x4;
typedef __attribute__((ext_vector_type(8)))  float  v8f;
typedef __attribute__((ext_vector_type(4)))  float  f32x4;

union Frag16 { f32x4 f4[2]; v16bf v; };

__device__ __forceinline__ __bf16 f2bf(float f) {
  unsigned u = __builtin_bit_cast(unsigned, f);
  u += 0x7FFFu + ((u >> 16) & 1u);            // round-to-nearest-even
  unsigned short h = (unsigned short)(u >> 16);
  return __builtin_bit_cast(__bf16, h);
}

// ---- gfx1250 async global->LDS staging (guarded; falls back to reg copy) ----
__device__ __forceinline__ void stage16(void* lds_dst, const void* gsrc) {
#if __has_builtin(__builtin_amdgcn_global_load_async_to_lds_b128)
  typedef int v4i __attribute__((vector_size(16)));
  typedef __attribute__((address_space(1))) void gvoid;
  typedef __attribute__((address_space(3))) void lvoid;
  typedef __attribute__((address_space(1))) v4i gv4i;
  typedef __attribute__((address_space(3))) v4i lv4i;
  __builtin_amdgcn_global_load_async_to_lds_b128(
      (gv4i*)(gvoid*)(void*)gsrc, (lv4i*)(lvoid*)lds_dst, 0, 0);
#else
  *(f32x4*)lds_dst = *(const f32x4*)gsrc;
#endif
}

__device__ __forceinline__ void stage_wait() {
#if __has_builtin(__builtin_amdgcn_global_load_async_to_lds_b128)
#if __has_builtin(__builtin_amdgcn_s_wait_asynccnt)
  __builtin_amdgcn_s_wait_asynccnt(0);
#else
  asm volatile("s_wait_asynccnt 0x0" ::: "memory");
#endif
#endif
}

__global__ void convert_f32_to_bf16(const float* __restrict__ in,
                                    __bf16* __restrict__ out, int n) {
  int i = (blockIdx.x * blockDim.x + threadIdx.x) * 4;
  if (i < n) {
    f32x4 v = *(const f32x4*)(in + i);
    bf16x4 o;
    o.x = f2bf(v.x); o.y = f2bf(v.y); o.z = f2bf(v.z); o.w = f2bf(v.w);
    *(bf16x4*)(out + i) = o;
  }
}

// Generic WMMA GEMM:  C = alpha * (A[MxK] @ B[NxK]^T) + bias
//   mode 0: f32 C, row-major ldc
//   mode 1: bf16 C, row-major ldc
//   mode 2: bf16 C stored transposed per batch: C[(m/seqS)*N + n][m%seqS]
__global__ __launch_bounds__(256, 2)
void gemm_wmma_bf16(const __bf16* __restrict__ A, const __bf16* __restrict__ Bm,
                    const float* __restrict__ bias, float alpha,
                    void* __restrict__ Cv,
                    int M, int N, int K, int lda, int ldb, int ldc,
                    long long aStrideZ, long long bStrideZ, long long cStrideZ,
                    int mode, int seqS) {
  __shared__ __align__(16) __bf16 As[2 * BM * LDSP];
  __shared__ __align__(16) __bf16 Bs[2 * BNT * LDSP];

  const int tid   = threadIdx.x;
  const int lane  = tid & 31;
  const int wid   = tid >> 5;
  const int waveM = wid & 3;    // 4 waves along M (32 rows each)
  const int waveN = wid >> 2;   // 2 waves along N (64 cols each)

  const int m0 = blockIdx.y * BM;
  const int n0 = blockIdx.x * BNT;
  const int z  = blockIdx.z;

  const __bf16* Ab = A  + (size_t)z * aStrideZ;
  const __bf16* Bb = Bm + (size_t)z * bStrideZ;

  v8f acc[2][4] = {};

  const int arow  = tid >> 1;          // 0..127: one row of A and of B
  const int ahalf = (tid & 1) * 32;    // half-row column offset (halves)
  const int nk    = K / BK;

  // Stage K-step kt into LDS buffer `buf` (A: 128x64, B: 128x64; 16B moves).
  auto stage_tiles = [&](int kt, int buf) {
    const int k0 = kt * BK;
    __bf16* Ad = As + buf * (BM * LDSP);
    __bf16* Bd = Bs + buf * (BNT * LDSP);
#pragma unroll
    for (int c = 0; c < 4; ++c) {
      const int col = ahalf + c * 8;
      stage16(&Ad[arow * LDSP + col],
              Ab + (size_t)(m0 + arow) * lda + k0 + col);
      stage16(&Bd[arow * LDSP + col],
              Bb + (size_t)(n0 + arow) * ldb + k0 + col);
    }
  };

  const int klo = (lane >> 4) << 3;   // A frag: lanes 16-31 hold K+8 pairs
  const int kb  = (lane >> 4) << 3;   // B frag: lanes 16-31 hold K+8 pairs

  // Two 32-deep WMMA sub-steps per staged 64-wide tile: 16 WMMAs / barrier.
  auto compute = [&](int buf) {
    const __bf16* Ac = As + buf * (BM * LDSP);
    const __bf16* Bc = Bs + buf * (BNT * LDSP);
#pragma unroll
    for (int ks = 0; ks < 2; ++ks) {
      const int kofs = ks * 32;
      Frag16 af[2], bfr[4];
#pragma unroll
      for (int i = 0; i < 2; ++i) {
        const __bf16* ap =
            &Ac[(waveM * 32 + i * 16 + (lane & 15)) * LDSP + kofs + klo];
        af[i].f4[0] = *(const f32x4*)ap;         // K = klo .. klo+7
        af[i].f4[1] = *(const f32x4*)(ap + 16);  // K = klo+16 .. klo+23
      }
#pragma unroll
      for (int j = 0; j < 4; ++j) {
        const __bf16* bp =
            &Bc[(waveN * 64 + j * 16 + (lane & 15)) * LDSP + kofs + kb];
        bfr[j].f4[0] = *(const f32x4*)bp;        // K = kb .. kb+7
        bfr[j].f4[1] = *(const f32x4*)(bp + 16); // K = kb+16 .. kb+23
      }
#pragma unroll
      for (int i = 0; i < 2; ++i)
#pragma unroll
        for (int j = 0; j < 4; ++j)
          acc[i][j] = __builtin_amdgcn_wmma_f32_16x16x32_bf16(
              false, af[i].v, false, bfr[j].v, (short)0, acc[i][j], false, false);
    }
  };

  stage_tiles(0, 0);
  stage_wait();
  __syncthreads();

  for (int kt = 0; kt < nk - 1; ++kt) {        // branch-free steady state
    stage_tiles(kt + 1, (kt + 1) & 1);
    compute(kt & 1);
    stage_wait();                               // our async LDS writes done...
    __syncthreads();                            // ...and visible to all waves
  }
  compute((nk - 1) & 1);                        // peeled last step

  // Store D fragments: VGPR r -> M = r (+8 for lanes 16-31), N = lane&15.
  const int mBase = m0 + waveM * 32 + ((lane >> 4) << 3);
  const int nLane = lane & 15;
#pragma unroll
  for (int i = 0; i < 2; ++i) {
#pragma unroll
    for (int j = 0; j < 4; ++j) {
      const int ng = n0 + waveN * 64 + j * 16 + nLane;
      const float badd = bias ? bias[ng] : 0.0f;
#pragma unroll
      for (int r = 0; r < 8; ++r) {
        const int mg = mBase + i * 16 + r;
        const float val = alpha * acc[i][j][r] + badd;
        if (mode == 0) {
          ((float*)Cv)[(size_t)z * cStrideZ + (size_t)mg * ldc + ng] = val;
        } else if (mode == 1) {
          ((__bf16*)Cv)[(size_t)z * cStrideZ + (size_t)mg * ldc + ng] = f2bf(val);
        } else {  // transposed per-batch store (for V^T)
          const int b = mg / seqS, s = mg % seqS;
          ((__bf16*)Cv)[((size_t)b * N + ng) * seqS + s] = f2bf(val);
        }
      }
    }
  }
}

__device__ __forceinline__ float wave_max(float v) {
#pragma unroll
  for (int m = 16; m > 0; m >>= 1) v = fmaxf(v, __shfl_xor(v, m, 32));
  return v;
}
__device__ __forceinline__ float wave_sum(float v) {
#pragma unroll
  for (int m = 16; m > 0; m >>= 1) v += __shfl_xor(v, m, 32);
  return v;
}

// Row softmax over f32 scores[row][0..S), rewritten in place as bf16 P with
// row stride 2*S bf16 elements (same byte stride as the f32 rows).
__global__ __launch_bounds__(256)
void softmax_rows(float* __restrict__ scores, int S) {
  __shared__ float red[8];
  const int row = blockIdx.x;
  const int tid = threadIdx.x;
  const int wv  = tid >> 5;
  float* rp = scores + (size_t)row * S;

  float v[8];
  float mx = -3.4e38f;
#pragma unroll
  for (int t = 0; t < 8; ++t) {            // all global reads happen here
    v[t] = rp[tid + t * 256];
    mx = fmaxf(mx, v[t]);
  }
  mx = wave_max(mx);
  if ((tid & 31) == 0) red[wv] = mx;
  __syncthreads();                          // reads complete before any write
#pragma unroll
  for (int w = 0; w < 8; ++w) mx = fmaxf(mx, red[w]);

  float sum = 0.f;
#pragma unroll
  for (int t = 0; t < 8; ++t) { v[t] = __expf(v[t] - mx); sum += v[t]; }
  sum = wave_sum(sum);
  __syncthreads();                          // red[] reuse
  if ((tid & 31) == 0) red[wv] = sum;
  __syncthreads();
  sum = 0.f;
#pragma unroll
  for (int w = 0; w < 8; ++w) sum += red[w];
  const float inv = 1.0f / sum;

  __bf16* op = (__bf16*)scores + (size_t)row * (2 * S);
#pragma unroll
  for (int t = 0; t < 8; ++t) op[tid + t * 256] = f2bf(v[t] * inv);
}

extern "C" void kernel_launch(void* const* d_in, const int* in_sizes, int n_in,
                              void* d_out, int out_size, void* d_ws, size_t ws_size,
                              hipStream_t stream) {
  const float* x  = (const float*)d_in[0];
  const float* Wq = (const float*)d_in[1];
  const float* bq = (const float*)d_in[2];
  const float* Wk = (const float*)d_in[3];
  const float* bk = (const float*)d_in[4];
  const float* Wv = (const float*)d_in[5];
  const float* bv = (const float*)d_in[6];
  float* out = (float*)d_out;

  // Workspace layout (needs ~134 MB):
  char* ws = (char*)d_ws;
  const size_t MB = 1ull << 20;
  __bf16* xb  = (__bf16*)(ws + 0);        // 16 MB
  __bf16* Wqb = (__bf16*)(ws + 16 * MB);  //  2 MB
  __bf16* Wkb = (__bf16*)(ws + 18 * MB);  //  2 MB
  __bf16* Wvb = (__bf16*)(ws + 20 * MB);  //  2 MB
  __bf16* Qb  = (__bf16*)(ws + 22 * MB);  // 16 MB
  __bf16* Kb  = (__bf16*)(ws + 38 * MB);  // 16 MB
  __bf16* Vt  = (__bf16*)(ws + 54 * MB);  // 16 MB  (V^T per batch: [b][d][s])
  float*  Sc  = (float*)(ws + 70 * MB);   // 64 MB  (scores; becomes bf16 P)

  const int E = EMB, S = SEQ, Bt = BATCH, M = MTOT;
  dim3 blk(256);

  // f32 -> bf16 conversions (vectorized x4)
  {
    int n = M * E;
    convert_f32_to_bf16<<<n / 4 / 256, 256, 0, stream>>>(x, xb, n);
    n = E * E;
    convert_f32_to_bf16<<<n / 4 / 256, 256, 0, stream>>>(Wq, Wqb, n);
    convert_f32_to_bf16<<<n / 4 / 256, 256, 0, stream>>>(Wk, Wkb, n);
    convert_f32_to_bf16<<<n / 4 / 256, 256, 0, stream>>>(Wv, Wvb, n);
  }

  // QKV projections: [8192x1024] @ [1024x1024]^T + bias
  dim3 gq(E / BNT, M / BM, 1);
  gemm_wmma_bf16<<<gq, blk, 0, stream>>>(xb, Wqb, bq, 1.0f, Qb,
      M, E, E, E, E, E, 0, 0, 0, /*mode=*/1, S);
  gemm_wmma_bf16<<<gq, blk, 0, stream>>>(xb, Wkb, bk, 1.0f, Kb,
      M, E, E, E, E, E, 0, 0, 0, /*mode=*/1, S);
  gemm_wmma_bf16<<<gq, blk, 0, stream>>>(xb, Wvb, bv, 1.0f, Vt,
      M, E, E, E, E, E, 0, 0, 0, /*mode=*/2, S);   // store V transposed

  // scores = (Q @ K^T) / sqrt(E), per batch (f32 out)
  dim3 gs(S / BNT, S / BM, Bt);
  gemm_wmma_bf16<<<gs, blk, 0, stream>>>(Qb, Kb, nullptr, 0.03125f, Sc,
      S, S, E, E, E, S,
      (long long)S * E, (long long)S * E, (long long)S * S, /*mode=*/0, S);

  // softmax rows; P (bf16) written in place over scores
  softmax_rows<<<M, 256, 0, stream>>>(Sc, S);

  // out = P @ V  (A = bf16 P with lda=2*S; B = V^T rows [d][s])
  dim3 go(E / BNT, S / BM, Bt);
  gemm_wmma_bf16<<<go, blk, 0, stream>>>((const __bf16*)Sc, Vt, nullptr, 1.0f, out,
      S, E, S, 2 * S, S, E,
      (long long)S * 2 * S, (long long)E * S, (long long)S * E, /*mode=*/0, S);
}
